// SA_GC_9251359555717
// MI455X (gfx1250) — matrix operand order repaired and checked by reading, not compiled, not run on hardware
//
#include <hip/hip_runtime.h>

typedef __attribute__((ext_vector_type(16))) _Float16 v16h;
typedef __attribute__((ext_vector_type(8)))  float    v8f;

constexpr int   kN  = 32, kC = 64, kT = 256, kV = 25, kHh = 8, kHD = 8, kO = 64;
constexpr float kScale = 0.35355339059327373f;   // 8^-0.5
constexpr float kEps   = 1e-5f;

// ---- CDNA5 WMMA fragment gathers (wave32, contiguous K order shared by A and B) ----
// A 16x32 f16: lane L holds row M = m0 + (L&15); element e -> K = hl*16 + e
__device__ __forceinline__ v16h ldsA(const _Float16* p, int ld, int m0, int k0, int lane) {
  const int m = m0 + (lane & 15), hl = lane >> 4;
  v16h a;
#pragma unroll
  for (int e = 0; e < 8; ++e) a[e] = p[m * ld + k0 + hl * 16 + e];
#pragma unroll
  for (int e = 0; e < 8; ++e) a[8 + e] = p[m * ld + k0 + hl * 16 + 8 + e];
  return a;
}
// B 32x16 f16: lane L holds col N = n0 + (L&15); element e -> K = hl*16 + e
__device__ __forceinline__ v16h ldsB(const _Float16* p, int ld, int k0, int n0, int lane) {
  const int n = n0 + (lane & 15), hl = lane >> 4;
  v16h b;
#pragma unroll
  for (int e = 0; e < 16; ++e) b[e] = p[(k0 + hl * 16 + e) * ld + n];
  return b;
}
__device__ __forceinline__ v8f wmma16(v16h a, v16h b, v8f c) {
  return __builtin_amdgcn_wmma_f32_16x16x32_f16(false, a, false, b, (short)0, c, false, false);
}

__global__ __launch_bounds__(32) void sagc_fused(
    const float* __restrict__ x,    const float* __restrict__ lng,  const float* __restrict__ lnb,
    const float* __restrict__ wqk,  const float* __restrict__ bqk,  const float* __restrict__ topo,
    const float* __restrict__ convw,const float* __restrict__ convb,
    const float* __restrict__ bng,  const float* __restrict__ bnb,
    const float* __restrict__ bnm,  const float* __restrict__ bnv,
    float* __restrict__ out) {
  __shared__ float    sX[kV][kC];        // x tile f32 (for LN stats)
  __shared__ _Float16 featH[32][kC];     // x tile f16, rows 25..31 zero
  __shared__ _Float16 ynH[32][kC];       // layernormed, f16
  __shared__ _Float16 qH[32][64];        // q  [v][h*8+d]
  __shared__ _Float16 kHs[32][64];       // k  [v][h*8+d]
  __shared__ float    sDots[32][32];     // attention logits / exp scratch
  __shared__ _Float16 attnH[32][32];     // softmax * topology, padded zero
  __shared__ _Float16 zH[32][64];        // per-head aggregate

  const int b = blockIdx.x;
  const int n = b / kT, t = b - n * kT;
  const int lane = threadIdx.x;
  const int col = lane & 15, hl = lane >> 4;
  const int xBase = (n * kC) * kT * kV + t * kV;   // + c*T*V + v

  // ---- stage x tile (f32 + f16), zero pads ----
  for (int i = lane; i < kV * kC; i += 32) {
    int c = i / kV, v = i - c * kV;
    float val = x[xBase + c * kT * kV + v];
    sX[v][c] = val;
    featH[v][c] = (_Float16)val;
  }
  for (int i = lane; i < (32 - kV) * kC; i += 32) {
    int r = kV + i / kC, c = i % kC;
    featH[r][c] = (_Float16)0.f;
    ynH[r][c]   = (_Float16)0.f;
  }
  __syncthreads();

  // ---- LayerNorm over C per joint (population variance) ----
  if (lane < kV) {
    float s = 0.f, ss = 0.f;
    for (int c = 0; c < kC; ++c) { float v = sX[lane][c]; s += v; ss += v * v; }
    float mu  = s * (1.f / kC);
    float var = ss * (1.f / kC) - mu * mu;
    float inv = rsqrtf(var + kEps);
    for (int c = 0; c < kC; ++c)
      ynH[lane][c] = (_Float16)(((sX[lane][c] - mu) * inv) * lng[c] + lnb[c]);
  }
  __syncthreads();

  // ---- GEMM1: qk = yn[32x64] @ w_qk[64x128] + b_qk ; split into q/k ----
  for (int nt = 0; nt < 8; ++nt) {
    v8f d0 = {}, d1 = {};
    const int nn = nt * 16 + col;
    for (int ks = 0; ks < 2; ++ks) {
      const int k0 = ks * 32;
      v16h bt;                                  // B from global (L2-resident)
#pragma unroll
      for (int e = 0; e < 16; ++e) bt[e] = (_Float16)wqk[(k0 + hl * 16 + e) * 128 + nn];
      v16h a0 = ldsA(&ynH[0][0], kC, 0,  k0, lane);
      v16h a1 = ldsA(&ynH[0][0], kC, 16, k0, lane);
      d0 = wmma16(a0, bt, d0);
      d1 = wmma16(a1, bt, d1);
    }
    const float bias = bqk[nn];
#pragma unroll
    for (int r = 0; r < 8; ++r) {
      const int m = r + 8 * hl;
      _Float16 h0 = (_Float16)(d0[r] + bias);
      _Float16 h1 = (_Float16)(d1[r] + bias);
      if (nn < 64) { qH[m][nn] = h0;        qH[16 + m][nn] = h1; }
      else         { kHs[m][nn - 64] = h0;  kHs[16 + m][nn - 64] = h1; }
    }
  }
  __syncthreads();

  v8f acc[2][4];
  {
    v8f vz = {};
#pragma unroll
    for (int i = 0; i < 2; ++i)
#pragma unroll
      for (int j = 0; j < 4; ++j) acc[i][j] = vz;
  }

  for (int h = 0; h < kHh; ++h) {
    // ---- dots = q_h @ k_h^T  (K=8, zero-padded to 32) ----
    v16h aq0, aq1, bk0, bk1;
#pragma unroll
    for (int e = 0; e < 16; ++e) {
      const int ka = hl * 16 + e;
      aq0[e] = (ka < kHD) ? qH[col][h * kHD + ka]      : (_Float16)0.f;
      aq1[e] = (ka < kHD) ? qH[16 + col][h * kHD + ka] : (_Float16)0.f;
      const int kb = hl * 16 + e;
      bk0[e] = (kb < kHD) ? kHs[col][h * kHD + kb]      : (_Float16)0.f;
      bk1[e] = (kb < kHD) ? kHs[16 + col][h * kHD + kb] : (_Float16)0.f;
    }
    v8f vz = {};
    v8f d00 = wmma16(aq0, bk0, vz);
    v8f d01 = wmma16(aq0, bk1, vz);
    v8f d10 = wmma16(aq1, bk0, vz);
    v8f d11 = wmma16(aq1, bk1, vz);
#pragma unroll
    for (int r = 0; r < 8; ++r) {
      const int m = r + 8 * hl;
      sDots[m][col]           = d00[r];
      sDots[m][16 + col]      = d01[r];
      sDots[16 + m][col]      = d10[r];
      sDots[16 + m][16 + col] = d11[r];
    }
    __syncthreads();

    // ---- softmax over j (rows u handled by lanes) × topology ----
    if (lane < kV) {
      float mx = -3.4e38f;
      for (int j = 0; j < kV; ++j) mx = fmaxf(mx, sDots[lane][j] * kScale);
      float sum = 0.f;
      for (int j = 0; j < kV; ++j) {
        float e = __expf(sDots[lane][j] * kScale - mx);
        sDots[lane][j] = e;
        sum += e;
      }
      const float inv = 1.f / sum;
      for (int j = 0; j < 32; ++j) {
        float a = (j < kV) ? sDots[lane][j] * inv * topo[(h * kV + lane) * kV + j] : 0.f;
        attnH[lane][j] = (_Float16)a;
      }
    } else {
      for (int j = 0; j < 32; ++j) attnH[lane][j] = (_Float16)0.f;
    }
    __syncthreads();

    // ---- z_h = A_h[32x32] @ feat[32x64] ----
    {
      v16h a0 = ldsA(&attnH[0][0], 32, 0,  0, lane);
      v16h a1 = ldsA(&attnH[0][0], 32, 16, 0, lane);
      for (int nt = 0; nt < 4; ++nt) {
        v16h bt = ldsB(&featH[0][0], kC, 0, nt * 16, lane);
        v8f e0 = wmma16(a0, bt, vz);
        v8f e1 = wmma16(a1, bt, vz);
#pragma unroll
        for (int r = 0; r < 8; ++r) {
          const int m = r + 8 * hl;
          zH[m][nt * 16 + col]      = (_Float16)e0[r];
          zH[16 + m][nt * 16 + col] = (_Float16)e1[r];
        }
      }
    }
    __syncthreads();

    // ---- acc += z_h[32x64] @ conv_w[h]^T[64x64]  (B[k=c][n=o] = conv_w[h][o][c]) ----
    for (int nt = 0; nt < 4; ++nt) {
      const int o = nt * 16 + col;
      for (int ks = 0; ks < 2; ++ks) {
        const int k0 = ks * 32;
        v16h bt;
#pragma unroll
        for (int e = 0; e < 16; ++e)
          bt[e] = (_Float16)convw[(h * kO + o) * kC + k0 + hl * 16 + e];
        v16h a0 = ldsA(&zH[0][0], kC, 0,  k0, lane);
        v16h a1 = ldsA(&zH[0][0], kC, 16, k0, lane);
        acc[0][nt] = wmma16(a0, bt, acc[0][nt]);
        acc[1][nt] = wmma16(a1, bt, acc[1][nt]);
      }
    }
    __syncthreads();
  }

  // ---- epilogue: + sum_h conv_b, BN (inference), residual, ReLU, scatter [N,O,T,V] ----
  for (int nt = 0; nt < 4; ++nt) {
    const int o = nt * 16 + col;
    float cb = 0.f;
#pragma unroll
    for (int h = 0; h < kHh; ++h) cb += convb[h * kO + o];
    const float gsc = bng[o] * rsqrtf(bnv[o] + kEps);
    const float gsh = bnb[o] - bnm[o] * gsc;
#pragma unroll
    for (int mt = 0; mt < 2; ++mt) {
#pragma unroll
      for (int r = 0; r < 8; ++r) {
        const int u = mt * 16 + r + 8 * hl;
        if (u < kV) {
          float val = (acc[mt][nt][r] + cb) * gsc + gsh;
          val += x[xBase + o * kT * kV + u];     // residual (C == O)
          val = fmaxf(val, 0.f);
          out[((n * kO + o) * kT + t) * kV + u] = val;
        }
      }
    }
  }
}

extern "C" void kernel_launch(void* const* d_in, const int* in_sizes, int n_in,
                              void* d_out, int out_size, void* d_ws, size_t ws_size,
                              hipStream_t stream) {
  (void)in_sizes; (void)n_in; (void)out_size; (void)d_ws; (void)ws_size;
  const float* x     = (const float*)d_in[0];
  const float* lng   = (const float*)d_in[1];
  const float* lnb   = (const float*)d_in[2];
  const float* wqk   = (const float*)d_in[3];
  const float* bqk   = (const float*)d_in[4];
  const float* topo  = (const float*)d_in[5];
  const float* convw = (const float*)d_in[6];
  const float* convb = (const float*)d_in[7];
  const float* bng   = (const float*)d_in[8];
  const float* bnb   = (const float*)d_in[9];
  const float* bnm   = (const float*)d_in[10];
  const float* bnv   = (const float*)d_in[11];
  sagc_fused<<<kN * kT, 32, 0, stream>>>(x, lng, lnb, wqk, bqk, topo, convw, convb,
                                         bng, bnb, bnm, bnv, (float*)d_out);
}
